// GeneralConv_42666205118902
// MI455X (gfx1250) — compile-verified, matrix operand order checked by reading, not run on hardware
//
#include <hip/hip_runtime.h>

typedef _Float16 half_t;
typedef __attribute__((ext_vector_type(16))) _Float16 v16h;
typedef __attribute__((ext_vector_type(8)))  _Float16 v8h;
typedef __attribute__((ext_vector_type(8)))  float    v8f;

#define WAVES 8   // 8 waves * 16 rows = 128 edges/nodes per workgroup

__device__ __forceinline__ float gelu_f(float x) {
    return 0.5f * x * (1.0f + erff(x * 0.70710678118654752f));
}

__device__ __forceinline__ void atomicMaxF(float* a, float v) {
    if (v >= 0.0f) atomicMax((int*)a, __float_as_int(v));
    else           atomicMin((unsigned int*)a, __float_as_uint(v));
}

struct F2 { v8h lo, hi; };

// Load a 16x32 f16 A/B fragment chunk from a row-major 64-wide (or padded) row.
// lane 0-15 (half=0): K = ks+0..7, ks+16..23 ; lane 16-31 (half=1): K = ks+8..15, ks+24..31
template <typename P>
__device__ __forceinline__ v16h loadFrag(const P* rowBase, int ks, int half) {
    F2 f;
    f.lo = *(const v8h*)(rowBase + ks + half * 8);
    f.hi = *(const v8h*)(rowBase + ks + 16 + half * 8);
    return __builtin_bit_cast(v16h, f);
}

__device__ __forceinline__ v8f wmma32(v16h a, v16h b, v8f c) {
    return __builtin_amdgcn_wmma_f32_16x16x32_f16(false, a, false, b, (short)0, c,
                                                  false, false);
}

// ---------------- prep: weights -> f16 copies in ws ----------------
__global__ void k_prep(const float* Wt, const float* Wq, const float* Wks,
                       const float* Wkd, const float* Wvs, const float* Wvd,
                       half_t* Wt16, half_t* Wq16, half_t* Wks16, half_t* Wkd16,
                       half_t* Wvs16, half_t* Wvd16, float* wtl) {
    for (int i = threadIdx.x; i < 4096; i += 256) {
        int r = i >> 6, c = i & 63;
        Wt16[i]  = (half_t)Wt[r * 65 + c];   // drop last (time) column
        Wq16[i]  = (half_t)Wq[i];
        Wks16[i] = (half_t)Wks[i];
        Wkd16[i] = (half_t)Wkd[i];
        Wvs16[i] = (half_t)Wvs[i];
        Wvd16[i] = (half_t)Wvd[i];
    }
    if (threadIdx.x < 64) wtl[threadIdx.x] = Wt[threadIdx.x * 65 + 64];
}

// ---------------- init: m = -FLT_MAX, denom = 0, aggr = 0 ----------------
__global__ void k_init(float* mArr, float* denom, float* aggr, int N) {
    long i = (long)blockIdx.x * 256 + threadIdx.x;
    if (i < (long)N * 64) aggr[i] = 0.0f;
    if (i < N) { mArr[i] = -3.402823466e38f; denom[i] = 0.0f; }
}

// ---------------- LayerNorm: x[N,64] -> xn16[N,64] (f16) ----------------
__global__ __launch_bounds__(256) void k_ln(const float* x, const float* g,
                                            const float* b, half_t* xn16, int N) {
    int wv = threadIdx.x >> 5, lane = threadIdx.x & 31;
    long node = (long)blockIdx.x * WAVES + wv;
    if (node >= N) return;
    const float* p = x + node * 64;
    float v0 = p[lane], v1 = p[lane + 32];
    float s = v0 + v1, ss = v0 * v0 + v1 * v1;
    for (int off = 1; off < 32; off <<= 1) {
        s  += __shfl_xor(s, off);
        ss += __shfl_xor(ss, off);
    }
    float mu  = s * (1.0f / 64.0f);
    float var = ss * (1.0f / 64.0f) - mu * mu;
    float inv = rsqrtf(var + 1e-5f);
    xn16[node * 64 + lane]      = (half_t)((v0 - mu) * inv * g[lane] + b[lane]);
    xn16[node * 64 + lane + 32] = (half_t)((v1 - mu) * inv * g[lane + 32] + b[lane + 32]);
}

// ---------------- per-node q = xn @ Wq^T + bq (WMMA) ----------------
__global__ __launch_bounds__(256) void k_q(const half_t* xn16, const half_t* Wq16,
                                           const float* bq, float* qn, int N) {
    __shared__ __align__(16) half_t sW[64 * 72];
    for (int c = threadIdx.x; c < 512; c += 256) {
        int row = c >> 3, col = (c & 7) * 8;
        *(v8h*)&sW[row * 72 + col] = *(const v8h*)&Wq16[row * 64 + col];
    }
    __syncthreads();
    int lane = threadIdx.x & 31, wv = threadIdx.x >> 5;
    int half = lane >> 4, m = lane & 15;
    long base = ((long)blockIdx.x * WAVES + wv) * 16;
    if (base >= N) return;
    long row = base + m; if (row > N - 1) row = N - 1;
    const half_t* pA = xn16 + (size_t)row * 64;
    v16h a0 = loadFrag(pA, 0, half), a1 = loadFrag(pA, 32, half);
    v8f acc[4] = {};
    for (int t = 0; t < 4; t++) acc[t] = wmma32(a0, loadFrag(&sW[(t * 16 + m) * 72], 0,  half), acc[t]);
    for (int t = 0; t < 4; t++) acc[t] = wmma32(a1, loadFrag(&sW[(t * 16 + m) * 72], 32, half), acc[t]);
    for (int t = 0; t < 4; t++) {
        int n = m + 16 * t;
        float bqv = bq[n];
        for (int r = 0; r < 8; r++) {
            long node = base + r + 8 * half;
            if (node < N) qn[node * 64 + n] = acc[t][r] + bqv;
        }
    }
}

// ---------------- edge pass A: xt, k, att, segment-max ----------------
__global__ __launch_bounds__(256) void k_edgeA(
    const int* ei, const float* et, const float* es,
    const half_t* xn16, const float* qn,
    const half_t* Wt16, const float* wtl, const float* bt,
    const half_t* Wks16, const float* bks,
    const half_t* Wkd16, const float* bkd,
    half_t* xtg, float* att, float* mArr, int E) {
    __shared__ __align__(16) half_t sWt[64 * 72];
    __shared__ __align__(16) half_t sWks[64 * 72];
    __shared__ __align__(16) half_t sWkd[64 * 72];
    __shared__ __align__(16) half_t sXT[WAVES * 16 * 72];
    for (int c = threadIdx.x; c < 512; c += 256) {
        int row = c >> 3, col = (c & 7) * 8;
        *(v8h*)&sWt[row * 72 + col]  = *(const v8h*)&Wt16[row * 64 + col];
        *(v8h*)&sWks[row * 72 + col] = *(const v8h*)&Wks16[row * 64 + col];
        *(v8h*)&sWkd[row * 72 + col] = *(const v8h*)&Wkd16[row * 64 + col];
    }
    __syncthreads();

    int lane = threadIdx.x & 31, wv = threadIdx.x >> 5;
    int half = lane >> 4, m = lane & 15;
    long base = ((long)blockIdx.x * WAVES + wv) * 16;
    if (base >= E) return;

    long eL = base + m; if (eL > E - 1) eL = E - 1;
    int   srcI = ei[eL];
    int   dstI = ei[(long)E + eL];
    float tL   = et[eL];
    float esL  = es[eL];

    int dstEr[8]; float esEr[8], tEr[8];
    for (int r = 0; r < 8; r++) {
        int sl = r + 8 * half;
        dstEr[r] = __shfl(dstI, sl);
        esEr[r]  = __shfl(esL, sl);
        tEr[r]   = __shfl(tL, sl);
    }

    // --- xt = gelu(x_j @ Wt64^T + t*wtl + bt) + te ---
    const half_t* pA = xn16 + (size_t)srcI * 64;
    v16h a0 = loadFrag(pA, 0, half), a1 = loadFrag(pA, 32, half);
    v8f acc[4] = {};
    for (int t = 0; t < 4; t++) acc[t] = wmma32(a0, loadFrag(&sWt[(t * 16 + m) * 72], 0,  half), acc[t]);
    for (int t = 0; t < 4; t++) acc[t] = wmma32(a1, loadFrag(&sWt[(t * 16 + m) * 72], 32, half), acc[t]);

    half_t* myXT = &sXT[wv * 16 * 72];
    for (int t = 0; t < 4; t++) {
        int n = m + 16 * t;
        float bias = bt[n], wl = wtl[n];
        float div = __expf(-(float)(2 * (n >> 1)) * (9.210340372f / 64.0f));
        for (int r = 0; r < 8; r++) {
            float val = acc[t][r] + bias + tEr[r] * wl;
            val = gelu_f(val);
            float ang = tEr[r] * 200.0f * div;
            val += (n & 1) ? __cosf(ang) : __sinf(ang);
            myXT[(r + 8 * half) * 72 + n] = (half_t)val;
        }
    }
    __builtin_amdgcn_wave_barrier();
    asm volatile("s_wait_dscnt 0" ::: "memory");
    __builtin_amdgcn_wave_barrier();

    // stage xt to global (f16) for pass B
    half_t* gXT = xtg + (size_t)base * 64;
    for (int c = lane; c < 128; c += 32) {
        int row = c >> 3, col = (c & 7) * 8;
        if (base + row < E)
            *(v8h*)&gXT[row * 64 + col] = *(const v8h*)&myXT[row * 72 + col];
    }

    // --- k_s / k_d GEMMs from LDS xt ---
    const half_t* pX = &myXT[m * 72];
    v16h xa0 = loadFrag(pX, 0, half), xa1 = loadFrag(pX, 32, half);
    v8f aS[4] = {}, aD[4] = {};
    for (int t = 0; t < 4; t++) {
        aS[t] = wmma32(xa0, loadFrag(&sWks[(t * 16 + m) * 72], 0, half), aS[t]);
        aD[t] = wmma32(xa0, loadFrag(&sWkd[(t * 16 + m) * 72], 0, half), aD[t]);
    }
    for (int t = 0; t < 4; t++) {
        aS[t] = wmma32(xa1, loadFrag(&sWks[(t * 16 + m) * 72], 32, half), aS[t]);
        aD[t] = wmma32(xa1, loadFrag(&sWkd[(t * 16 + m) * 72], 32, half), aD[t]);
    }

    // --- att = (k . q[dst]) / 8 ---
    float accA[8] = {};
    for (int t = 0; t < 4; t++) {
        int n = m + 16 * t;
        float bs = bks[n], bd = bkd[n];
        for (int r = 0; r < 8; r++) {
            float ks_ = aS[t][r] + bs, kd_ = aD[t][r] + bd;
            float kk = kd_ + esEr[r] * (ks_ - kd_);
            float qv = qn[(size_t)dstEr[r] * 64 + n];
            accA[r] += kk * qv;
        }
    }
    for (int r = 0; r < 8; r++)
        for (int off = 1; off < 16; off <<= 1)
            accA[r] += __shfl_xor(accA[r], off);

    if (m == 0) {
        for (int r = 0; r < 8; r++) {
            long e = base + r + 8 * half;
            if (e < E) {
                float a = accA[r] * 0.125f;
                att[e] = a;
                atomicMaxF(&mArr[dstEr[r]], a);
            }
        }
    }
}

// ---------------- edge pass B: v, exp, atomic scatter ----------------
__global__ __launch_bounds__(256) void k_edgeB(
    const int* ei, const float* es, const half_t* xtg,
    const half_t* Wvs16, const float* bvs,
    const half_t* Wvd16, const float* bvd,
    const float* att, const float* mArr,
    float* denom, float* aggr, int E) {
    __shared__ __align__(16) half_t sWs[64 * 72];
    __shared__ __align__(16) half_t sWd[64 * 72];
    for (int c = threadIdx.x; c < 512; c += 256) {
        int row = c >> 3, col = (c & 7) * 8;
        *(v8h*)&sWs[row * 72 + col] = *(const v8h*)&Wvs16[row * 64 + col];
        *(v8h*)&sWd[row * 72 + col] = *(const v8h*)&Wvd16[row * 64 + col];
    }
    __syncthreads();

    int lane = threadIdx.x & 31, wv = threadIdx.x >> 5;
    int half = lane >> 4, m = lane & 15;
    long base = ((long)blockIdx.x * WAVES + wv) * 16;
    if (base >= E) return;

    long eL = base + m; if (eL > E - 1) eL = E - 1;
    int   dstI = ei[(long)E + eL];
    float esL  = es[eL];
    float exL  = __expf(att[eL] - mArr[dstI]);

    int dstEr[8]; float esEr[8], exEr[8];
    for (int r = 0; r < 8; r++) {
        int sl = r + 8 * half;
        dstEr[r] = __shfl(dstI, sl);
        esEr[r]  = __shfl(esL, sl);
        exEr[r]  = __shfl(exL, sl);
    }

    if (m == 0) {
        for (int r = 0; r < 8; r++) {
            long e = base + r + 8 * half;
            if (e < E) atomicAdd(&denom[dstEr[r]], exEr[r]);
        }
    }

    const half_t* pX = xtg + (size_t)eL * 64;
    v16h xa0 = loadFrag(pX, 0, half), xa1 = loadFrag(pX, 32, half);
    v8f aS[4] = {}, aD[4] = {};
    for (int t = 0; t < 4; t++) {
        aS[t] = wmma32(xa0, loadFrag(&sWs[(t * 16 + m) * 72], 0, half), aS[t]);
        aD[t] = wmma32(xa0, loadFrag(&sWd[(t * 16 + m) * 72], 0, half), aD[t]);
    }
    for (int t = 0; t < 4; t++) {
        aS[t] = wmma32(xa1, loadFrag(&sWs[(t * 16 + m) * 72], 32, half), aS[t]);
        aD[t] = wmma32(xa1, loadFrag(&sWd[(t * 16 + m) * 72], 32, half), aD[t]);
    }

    for (int t = 0; t < 4; t++) {
        int n = m + 16 * t;
        float bs = bvs[n], bd = bvd[n];
        for (int r = 0; r < 8; r++) {
            long e = base + r + 8 * half;
            if (e < E) {
                float vs = aS[t][r] + bs, vd = aD[t][r] + bd;
                float vv = vd + esEr[r] * (vs - vd);
                atomicAdd(&aggr[(size_t)dstEr[r] * 64 + n], vv * exEr[r]);
            }
        }
    }
}

// ---------------- finalize: out = x + gelu(aggr / (denom + 1e-16)) ----------------
__global__ void k_fin(const float* x, const float* aggr, const float* denom,
                      float* out, long NB) {
    long i = (long)blockIdx.x * 256 + threadIdx.x;
    if (i < NB) {
        float d = denom[i >> 6];
        float a = aggr[i] / (d + 1e-16f);
        out[i] = x[i] + gelu_f(a);
    }
}

extern "C" void kernel_launch(void* const* d_in, const int* in_sizes, int n_in,
                              void* d_out, int out_size, void* d_ws, size_t ws_size,
                              hipStream_t stream) {
    const float* x     = (const float*)d_in[0];
    const int*   ei    = (const int*)d_in[1];
    const float* et    = (const float*)d_in[2];
    const float* esm   = (const float*)d_in[4];
    const float* gam   = (const float*)d_in[5];
    const float* bet   = (const float*)d_in[6];
    const float* Wt    = (const float*)d_in[7];
    const float* bt    = (const float*)d_in[8];
    const float* Wq    = (const float*)d_in[9];
    const float* bq    = (const float*)d_in[10];
    const float* Wks   = (const float*)d_in[11];
    const float* bks   = (const float*)d_in[12];
    const float* Wkd   = (const float*)d_in[13];
    const float* bkd   = (const float*)d_in[14];
    const float* Wvs   = (const float*)d_in[15];
    const float* bvs   = (const float*)d_in[16];
    const float* Wvd   = (const float*)d_in[17];
    const float* bvd   = (const float*)d_in[18];

    int N = in_sizes[0] / 64;
    int E = in_sizes[2];
    long NB = (long)N * 64;

    char* w = (char*)d_ws;
    size_t off = 0;
    auto take = [&](size_t bytes) -> void* {
        void* p = w + off;
        off = (off + bytes + 255) & ~(size_t)255;
        return p;
    };
    half_t* xn16  = (half_t*)take((size_t)NB * 2);
    float*  qn    = (float*)take((size_t)NB * 4);
    half_t* Wt16  = (half_t*)take(8192);
    half_t* Wq16  = (half_t*)take(8192);
    half_t* Wks16 = (half_t*)take(8192);
    half_t* Wkd16 = (half_t*)take(8192);
    half_t* Wvs16 = (half_t*)take(8192);
    half_t* Wvd16 = (half_t*)take(8192);
    float*  wtl   = (float*)take(256);
    float*  att   = (float*)take((size_t)E * 4);
    float*  mArr  = (float*)take((size_t)N * 4);
    float*  denom = (float*)take((size_t)N * 4);
    float*  aggr  = (float*)take((size_t)NB * 4);
    half_t* xtg   = (half_t*)take((size_t)E * 64 * 2);
    (void)ws_size; (void)n_in; (void)out_size;

    k_prep<<<1, 256, 0, stream>>>(Wt, Wq, Wks, Wkd, Wvs, Wvd,
                                  Wt16, Wq16, Wks16, Wkd16, Wvs16, Wvd16, wtl);
    k_init<<<(int)((NB + 255) / 256), 256, 0, stream>>>(mArr, denom, aggr, N);
    k_ln<<<(N + WAVES - 1) / WAVES, 256, 0, stream>>>(x, gam, bet, xn16, N);
    k_q<<<(N + 127) / 128, 256, 0, stream>>>(xn16, Wq16, bq, qn, N);
    k_edgeA<<<(E + 127) / 128, 256, 0, stream>>>(ei, et, esm, xn16, qn,
                                                 Wt16, wtl, bt, Wks16, bks,
                                                 Wkd16, bkd, xtg, att, mArr, E);
    k_edgeB<<<(E + 127) / 128, 256, 0, stream>>>(ei, esm, xtg, Wvs16, bvs,
                                                 Wvd16, bvd, att, mArr,
                                                 denom, aggr, E);
    k_fin<<<(int)((NB + 255) / 256), 256, 0, stream>>>(x, aggr, denom,
                                                       (float*)d_out, NB);
}
